// LASAGE_S_89601607729377
// MI455X (gfx1250) — matrix operand order repaired and checked by reading, not compiled, not run on hardware
//
#include <hip/hip_runtime.h>
#include <hip/hip_bf16.h>

typedef __attribute__((ext_vector_type(16))) _Float16 v16h;
typedef __attribute__((ext_vector_type(8)))  float    v8f;

#define D_IN  128
#define D_HID 32
#define D_OUT 64

// ---------------------------------------------------------------------------
// Pre-swizzle a weight matrix W[Nout][K] (row-major, f32) into per-lane WMMA
// B-fragments (f16).  Fragment order: ((kk*ntiles + nt)*32 + lane)*16 + e
// B layout (16-bit, 32x16 KxN block): col = nt*16 + (lane&15),
//                                     k   = kk*32 + (lane<16 ? 0 : 8) + e + (e<8 ? 0 : 8)
// ---------------------------------------------------------------------------
__global__ void prep_wt(const float* __restrict__ W, _Float16* __restrict__ B,
                        int K, int Nout, int ksteps, int ntiles) {
    int idx = blockIdx.x * blockDim.x + threadIdx.x;
    int total = ksteps * ntiles * 32 * 16;
    if (idx >= total) return;
    int e    = idx & 15;
    int lane = (idx >> 4) & 31;
    int t    = idx >> 9;              // kk*ntiles + nt
    int nt   = t % ntiles;
    int kk   = t / ntiles;
    int col  = nt * 16 + (lane & 15);
    int k    = kk * 32 + ((lane < 16) ? 0 : 8) + e + ((e < 8) ? 0 : 8);
    B[idx] = (_Float16)W[col * K + k];
}

// summed0 = x (self loop), cnt = 1
__global__ void init0_kernel(const float* __restrict__ x,
                             float* __restrict__ summed0,
                             float* __restrict__ cnt, int N) {
    long long i = (long long)blockIdx.x * blockDim.x + threadIdx.x;
    long long tot = (long long)N * D_IN;
    if (i < tot) summed0[i] = x[i];
    if (i < N)   cnt[i] = 1.0f;
}

// scatter-add: summed[dst] += feat[src] ; optionally cnt[dst] += 1
__global__ void scatter_add(const int* __restrict__ src,
                            const int* __restrict__ dst,
                            const float* __restrict__ feat,
                            float* __restrict__ summed,
                            float* __restrict__ cnt,
                            int E, int D, int add_cnt) {
    long long tid = (long long)blockIdx.x * blockDim.x + threadIdx.x;
    int chunks = D >> 2;
    long long tot = (long long)E * chunks;
    if (tid >= tot) return;
    int e = (int)(tid / chunks);
    int c = (int)(tid % chunks) * 4;
    int s = src[e];
    int d = dst[e];
    float4 v = *(const float4*)(feat + (long long)s * D + c);
    float* p = summed + (long long)d * D + c;
    atomicAdd(p + 0, v.x);
    atomicAdd(p + 1, v.y);
    atomicAdd(p + 2, v.z);
    atomicAdd(p + 3, v.w);
    if (add_cnt && (tid % chunks) == 0) atomicAdd(cnt + d, 1.0f);
}

// ---------------------------------------------------------------------------
// Layer 0: per 16-node tile (one wave): fr/be GEMMs on aggr, self GEMM on x,
// sigmoid gate on x, gated relu combine, outer relu.  Writes h to ws.h,
// ws.summed1 (self-loop init for layer 1) and d_out (x32).
// ---------------------------------------------------------------------------
__global__ void __launch_bounds__(64)
gemm_layer0(const float* __restrict__ x, const float* __restrict__ summed,
            const float* __restrict__ cnt,
            const _Float16* __restrict__ Bfr, const _Float16* __restrict__ Bbe,
            const _Float16* __restrict__ Bself,
            const float* __restrict__ wbal, const float* __restrict__ bbal,
            const float* __restrict__ bfr, const float* __restrict__ bbe,
            const float* __restrict__ bself,
            float* __restrict__ h, float* __restrict__ summed1,
            float* __restrict__ x32out, int N, int ntiles_total) {
    const int lane = threadIdx.x & 31;
    const int wave = threadIdx.x >> 5;
    int tile = blockIdx.x * 2 + wave;
    if (tile >= ntiles_total) tile = ntiles_total - 1;   // full-EXEC clamp
    const int base = tile * 16;

    __shared__ float xs[2][16][132];
    __shared__ float ss[2][16][132];
    float (*X)[132] = xs[wave];
    float (*S)[132] = ss[wave];

    __builtin_prefetch(Bfr, 0, 1);

    // stage 16x128 tiles of x and summed into LDS (coalesced float4)
    for (int i = lane; i < 16 * 32; i += 32) {
        int r  = i >> 5;
        int c4 = (i & 31) * 4;
        int gr = min(base + r, N - 1);
        float4 vx = *(const float4*)(x      + (long long)gr * D_IN + c4);
        float4 vs = *(const float4*)(summed + (long long)gr * D_IN + c4);
        X[r][c4 + 0] = vx.x; X[r][c4 + 1] = vx.y; X[r][c4 + 2] = vx.z; X[r][c4 + 3] = vx.w;
        S[r][c4 + 0] = vs.x; S[r][c4 + 1] = vs.y; S[r][c4 + 2] = vs.z; S[r][c4 + 3] = vs.w;
    }
    __syncthreads();

    const int row     = lane & 15;
    const int halfsel = lane >> 4;
    const float invc  = 1.0f / cnt[min(base + row, N - 1)];

    // gate: bal = sigmoid(x_row . wbal + bbal); split 64+64 cols across halves
    float part = 0.0f;
    for (int c = halfsel * 64; c < halfsel * 64 + 64; ++c)
        part += X[row][c] * wbal[c];
    part += __shfl_xor(part, 16, 32);
    const float bal = 1.0f / (1.0f + __expf(-(part + bbal[0])));

    v8f accfr[2] = {};
    v8f accbe[2] = {};
    v8f accsf[2] = {};

    const int kb = (lane < 16) ? 0 : 8;
    #pragma unroll
    for (int kk = 0; kk < 4; ++kk) {
        v16h Aag, Ax;
        #pragma unroll
        for (int e2 = 0; e2 < 16; ++e2) {
            int k = kk * 32 + kb + e2 + ((e2 < 8) ? 0 : 8);
            Aag[e2] = (_Float16)(S[row][k] * invc);
            Ax[e2]  = (_Float16)(X[row][k]);
        }
        #pragma unroll
        for (int nt = 0; nt < 2; ++nt) {
            int off = ((kk * 2 + nt) * 32 + lane) * 16;
            v16h Bf = *(const v16h*)(Bfr   + off);
            v16h Bb = *(const v16h*)(Bbe   + off);
            v16h Bs = *(const v16h*)(Bself + off);
            accfr[nt] = __builtin_amdgcn_wmma_f32_16x16x32_f16(
                false, Aag, false, Bf, (short)0, accfr[nt], false, false);
            accbe[nt] = __builtin_amdgcn_wmma_f32_16x16x32_f16(
                false, Aag, false, Bb, (short)0, accbe[nt], false, false);
            accsf[nt] = __builtin_amdgcn_wmma_f32_16x16x32_f16(
                false, Ax,  false, Bs, (short)0, accsf[nt], false, false);
        }
    }

    // combine: out = self + bself + relu(fr*bal + be*(1-bal)); h = relu(out)
    #pragma unroll
    for (int nt = 0; nt < 2; ++nt) {
        int col = nt * 16 + row;
        float bf = bfr[col], bb = bbe[col], bs = bself[col];
        #pragma unroll
        for (int r = 0; r < 8; ++r) {
            int orow   = r + halfsel * 8;            // C/D row for this VGPR
            float balr = __shfl(bal, orow, 32);
            float fr   = accfr[nt][r] + bf;
            float be   = accbe[nt][r] + bb;
            float upd  = fmaxf(fr * balr + be * (1.0f - balr), 0.0f);
            float hv   = fmaxf(accsf[nt][r] + bs + upd, 0.0f);
            int node   = base + orow;
            if (node < N) {
                h[(long long)node * D_HID + col]       = hv;
                summed1[(long long)node * D_HID + col] = hv;
                x32out[(long long)node * D_HID + col]  = hv;
            }
        }
    }
}

// ---------------------------------------------------------------------------
// Layer 1: [N,32] -> [N,64], same structure, single K-step, 4 N-tiles,
// no outer relu.
// ---------------------------------------------------------------------------
__global__ void __launch_bounds__(64)
gemm_layer1(const float* __restrict__ hsrc, const float* __restrict__ summed,
            const float* __restrict__ cnt,
            const _Float16* __restrict__ Bfr, const _Float16* __restrict__ Bbe,
            const _Float16* __restrict__ Bself,
            const float* __restrict__ wbal, const float* __restrict__ bbal,
            const float* __restrict__ bfr, const float* __restrict__ bbe,
            const float* __restrict__ bself,
            float* __restrict__ out64, int N, int ntiles_total) {
    const int lane = threadIdx.x & 31;
    const int wave = threadIdx.x >> 5;
    int tile = blockIdx.x * 2 + wave;
    if (tile >= ntiles_total) tile = ntiles_total - 1;
    const int base = tile * 16;

    __shared__ float hs[2][16][36];
    __shared__ float ss[2][16][36];
    float (*H)[36] = hs[wave];
    float (*S)[36] = ss[wave];

    __builtin_prefetch(Bfr, 0, 1);

    for (int i = lane; i < 16 * 8; i += 32) {     // 16 rows x 8 float4 chunks
        int r  = i >> 3;
        int c4 = (i & 7) * 4;
        int gr = min(base + r, N - 1);
        float4 vh = *(const float4*)(hsrc   + (long long)gr * D_HID + c4);
        float4 vs = *(const float4*)(summed + (long long)gr * D_HID + c4);
        H[r][c4 + 0] = vh.x; H[r][c4 + 1] = vh.y; H[r][c4 + 2] = vh.z; H[r][c4 + 3] = vh.w;
        S[r][c4 + 0] = vs.x; S[r][c4 + 1] = vs.y; S[r][c4 + 2] = vs.z; S[r][c4 + 3] = vs.w;
    }
    __syncthreads();

    const int row     = lane & 15;
    const int halfsel = lane >> 4;
    const float invc  = 1.0f / cnt[min(base + row, N - 1)];

    float part = 0.0f;
    for (int c = halfsel * 16; c < halfsel * 16 + 16; ++c)
        part += H[row][c] * wbal[c];
    part += __shfl_xor(part, 16, 32);
    const float bal = 1.0f / (1.0f + __expf(-(part + bbal[0])));

    v8f accfr[4] = {};
    v8f accbe[4] = {};
    v8f accsf[4] = {};

    const int kb = (lane < 16) ? 0 : 8;
    v16h Aag, Ah;
    #pragma unroll
    for (int e2 = 0; e2 < 16; ++e2) {
        int k = kb + e2 + ((e2 < 8) ? 0 : 8);      // K = 0..31, single step
        Aag[e2] = (_Float16)(S[row][k] * invc);
        Ah[e2]  = (_Float16)(H[row][k]);
    }
    #pragma unroll
    for (int nt = 0; nt < 4; ++nt) {
        int off = (nt * 32 + lane) * 16;
        v16h Bf = *(const v16h*)(Bfr   + off);
        v16h Bb = *(const v16h*)(Bbe   + off);
        v16h Bs = *(const v16h*)(Bself + off);
        accfr[nt] = __builtin_amdgcn_wmma_f32_16x16x32_f16(
            false, Aag, false, Bf, (short)0, accfr[nt], false, false);
        accbe[nt] = __builtin_amdgcn_wmma_f32_16x16x32_f16(
            false, Aag, false, Bb, (short)0, accbe[nt], false, false);
        accsf[nt] = __builtin_amdgcn_wmma_f32_16x16x32_f16(
            false, Ah,  false, Bs, (short)0, accsf[nt], false, false);
    }

    #pragma unroll
    for (int nt = 0; nt < 4; ++nt) {
        int col = nt * 16 + row;
        float bf = bfr[col], bb = bbe[col], bs = bself[col];
        #pragma unroll
        for (int r = 0; r < 8; ++r) {
            int orow   = r + halfsel * 8;
            float balr = __shfl(bal, orow, 32);
            float fr   = accfr[nt][r] + bf;
            float be   = accbe[nt][r] + bb;
            float upd  = fmaxf(fr * balr + be * (1.0f - balr), 0.0f);
            float ov   = accsf[nt][r] + bs + upd;   // no outer relu
            int node   = base + orow;
            if (node < N)
                out64[(long long)node * D_OUT + col] = ov;
        }
    }
}

// ---------------------------------------------------------------------------
extern "C" void kernel_launch(void* const* d_in, const int* in_sizes, int n_in,
                              void* d_out, int out_size, void* d_ws, size_t ws_size,
                              hipStream_t stream) {
    const float* x    = (const float*)d_in[0];
    const int*   eidx = (const int*)d_in[1];
    // d_in[2], d_in[3]: dead W1/b1
    const float* Wfr0   = (const float*)d_in[4];
    const float* bfr0   = (const float*)d_in[5];
    const float* Wbe0   = (const float*)d_in[6];
    const float* bbe0   = (const float*)d_in[7];
    const float* Wbal0  = (const float*)d_in[8];
    const float* bbal0  = (const float*)d_in[9];
    const float* Wself0 = (const float*)d_in[10];
    const float* bself0 = (const float*)d_in[11];
    const float* Wfr1   = (const float*)d_in[12];
    const float* bfr1   = (const float*)d_in[13];
    const float* Wbe1   = (const float*)d_in[14];
    const float* bbe1   = (const float*)d_in[15];
    const float* Wbal1  = (const float*)d_in[16];
    const float* bbal1  = (const float*)d_in[17];
    const float* Wself1 = (const float*)d_in[18];
    const float* bself1 = (const float*)d_in[19];

    const int N = in_sizes[0] / D_IN;
    const int E = in_sizes[1] / 2;
    const int* src = eidx;
    const int* dst = eidx + E;

    // workspace carve-up (offsets in floats, padded to 64-float alignment)
    float* ws = (float*)d_ws;
    auto pad64 = [](size_t v) { return (v + 63) & ~(size_t)63; };
    size_t off = 0;
    float* summed0 = ws + off; off = pad64(off + (size_t)N * D_IN);
    float* cnt     = ws + off; off = pad64(off + (size_t)N);
    float* hbuf    = ws + off; off = pad64(off + (size_t)N * D_HID);
    float* summed1 = ws + off; off = pad64(off + (size_t)N * D_HID);
    _Float16* wtb  = (_Float16*)(ws + off);
    _Float16* Bfr0s   = wtb;            // 4*2*32*16 = 4096 halfs each (layer 0)
    _Float16* Bbe0s   = wtb + 4096;
    _Float16* Bself0s = wtb + 8192;
    _Float16* Bfr1s   = wtb + 12288;    // 1*4*32*16 = 2048 halfs each (layer 1)
    _Float16* Bbe1s   = wtb + 14336;
    _Float16* Bself1s = wtb + 16384;

    float* x32out = (float*)d_out;                       // [N,32]
    float* out64  = (float*)d_out + (size_t)N * D_HID;   // [N,64]

    // 1) swizzle weights into WMMA B-fragment order (f16)
    prep_wt<<<16, 256, 0, stream>>>(Wfr0,   Bfr0s,   D_IN,  D_HID, 4, 2);
    prep_wt<<<16, 256, 0, stream>>>(Wbe0,   Bbe0s,   D_IN,  D_HID, 4, 2);
    prep_wt<<<16, 256, 0, stream>>>(Wself0, Bself0s, D_IN,  D_HID, 4, 2);
    prep_wt<<<8,  256, 0, stream>>>(Wfr1,   Bfr1s,   D_HID, D_OUT, 1, 4);
    prep_wt<<<8,  256, 0, stream>>>(Wbe1,   Bbe1s,   D_HID, D_OUT, 1, 4);
    prep_wt<<<8,  256, 0, stream>>>(Wself1, Bself1s, D_HID, D_OUT, 1, 4);

    // 2) layer-0 aggregation: summed0 = x (self loop) + scatter(x[src]); cnt = deg+1
    {
        long long tot = (long long)N * D_IN;
        unsigned int blocks = (unsigned int)((tot + 255) / 256);
        init0_kernel<<<blocks, 256, 0, stream>>>(x, summed0, cnt, N);
    }
    {
        long long tot = (long long)E * (D_IN / 4);
        unsigned int blocks = (unsigned int)((tot + 255) / 256);
        scatter_add<<<blocks, 256, 0, stream>>>(src, dst, x, summed0, cnt, E, D_IN, 1);
    }

    // 3) layer-0 fused GEMM/gate/relu (WMMA); also seeds summed1 = h
    const int tiles = (N + 15) / 16;
    {
        unsigned int blocks = (unsigned int)((tiles + 1) / 2);
        gemm_layer0<<<blocks, 64, 0, stream>>>(
            x, summed0, cnt, Bfr0s, Bbe0s, Bself0s,
            Wbal0, bbal0, bfr0, bbe0, bself0,
            hbuf, summed1, x32out, N, tiles);
    }

    // 4) layer-1 aggregation: summed1 += scatter(h[src]) (cnt reused)
    {
        long long tot = (long long)E * (D_HID / 4);
        unsigned int blocks = (unsigned int)((tot + 255) / 256);
        scatter_add<<<blocks, 256, 0, stream>>>(src, dst, hbuf, summed1, cnt, E, D_HID, 0);
    }

    // 5) layer-1 fused GEMM/gate (WMMA), writes final [N,64]
    {
        unsigned int blocks = (unsigned int)((tiles + 1) / 2);
        gemm_layer1<<<blocks, 64, 0, stream>>>(
            hbuf, summed1, cnt, Bfr1s, Bbe1s, Bself1s,
            Wbal1, bbal1, bfr1, bbe1, bself1,
            out64, N, tiles);
    }
}
